// BNN_Fast_90202903151012
// MI455X (gfx1250) — matrix operand order repaired and checked by reading, not compiled, not run on hardware
//
#include <hip/hip_runtime.h>

// Problem dims (from reference)
#define NPTS 1024
#define WENS 256
#define XDIM 32
#define YDIM 32
#define HDIM 256
#define MT   64                 // rows of N per workgroup
#define KP   32                 // K panel (= WMMA bf16 K)
#define NTHREADS 256
#define PSZ  (HDIM * KP)        // 8192 bf16 per trunk panel (16 KB)
#define HPSZ (YDIM * KP)        // 1024 bf16 per head panel (2 KB)

typedef __attribute__((ext_vector_type(16))) __bf16 v16bf;
typedef __attribute__((ext_vector_type(8)))  float  v8f;

union ABFrag { v16bf v; unsigned u[8]; };

__device__ __forceinline__ unsigned short f2bf(float f) {
  unsigned u = __float_as_uint(f);
  u += 0x7FFFu + ((u >> 16) & 1u);       // round-to-nearest-even
  return (unsigned short)(u >> 16);
}

// Low 32 bits of a flat LDS address == wave-relative LDS byte offset (ISA 10.2)
__device__ __forceinline__ unsigned lds_off(const void* p) {
  return (unsigned)(unsigned long long)p;
}

// GVS-form async copy: LDS[loff] <- MEM[sbase + goff], 16 bytes per lane (ASYNCcnt)
__device__ __forceinline__ void async_b128(unsigned loff, unsigned goff, const void* sbase) {
  asm volatile("global_load_async_to_lds_b128 %0, %1, %2"
               :: "v"(loff), "v"(goff), "s"(sbase)
               : "memory");
}

__device__ __forceinline__ void wait_async0() {
#if __has_builtin(__builtin_amdgcn_s_wait_asynccnt)
  __builtin_amdgcn_s_wait_asynccnt(0);
#else
  asm volatile("s_wait_asynccnt 0" ::: "memory");
#endif
}

// Workgroup-cooperative async memcpy global->LDS (nshorts*2 must be multiple of 16)
__device__ __forceinline__ void issue_copy(const unsigned short* g, unsigned short* lds,
                                           int nshorts, int tid) {
  const int nchunk = (nshorts * 2) >> 4;
  const unsigned lbase = lds_off(lds);
  for (int c = tid; c < nchunk; c += NTHREADS)
    async_b128(lbase + (unsigned)(c * 16), (unsigned)(c * 16), g);
}

// A fragment: contiguous K order k = 16*half + s (matches B; K pairs contiguous -> 32-bit LDS loads)
__device__ __forceinline__ v16bf load_afrag(const unsigned short* actRow, int half) {
  ABFrag a;
#pragma unroll
  for (int p = 0; p < 8; ++p) {
    int off = half * 16 + 2 * p;
    a.u[p] = *(const unsigned*)(actRow + off);
  }
  return a.v;
}

// B fragment from LDS weights stored transposed [n][KP]: 32B contiguous per lane
__device__ __forceinline__ v16bf load_bfrag(const unsigned short* wtCol, int half) {
  ABFrag b;
  const unsigned short* base = wtCol + half * 16;
#pragma unroll
  for (int p = 0; p < 8; ++p) b.u[p] = *(const unsigned*)(base + 2 * p);
  return b.v;
}

__device__ __forceinline__ v8f wmma_bf16(v16bf a, v16bf b, v8f c) {
  return __builtin_amdgcn_wmma_f32_16x16x32_bf16(false, a, false, b, (short)0, c, false, false);
}

__device__ __forceinline__ float elu(float v) {
  return (v > 0.0f) ? v : (__expf(v) - 1.0f);   // branchless v_exp_f32 path
}

// ---------------- weight pre-conversion: fp32 [W][K][N] -> bf16 panels [W][K/32][N][32] -------------
template<int LOGN>
__global__ void __launch_bounds__(NTHREADS)
convert_weights(const float* __restrict__ src, unsigned short* __restrict__ dst, int nkp) {
  const int N = 1 << LOGN;
  const int blk = blockIdx.x;
  const int kp = blk % nkp;
  const int w  = blk / nkp;
  const float* s = src + (((size_t)w * nkp + kp) * KP) * N;
  unsigned short* d = dst + (size_t)blk * (KP << LOGN);
  for (int i = threadIdx.x; i < (KP << LOGN); i += NTHREADS) {
    int k = i >> LOGN;
    int n = i & (N - 1);
    d[(size_t)n * KP + k] = f2bf(s[(size_t)k * N + n]);   // transpose to [n][k]
  }
}

// ---------------- PRE path trunk layer: async double-buffered bf16 panels ----------------
// Per wave: mt in 0..3 (16-row M tile), nq in 0..1 (128-col N half -> 8 tiles)
__device__ __forceinline__ void dense_layer_pre(
    const unsigned short* actIn, int inStride, int ktot,
    const unsigned short* gWblk, const float* __restrict__ gB, bool elu_act,
    unsigned short* actOut, unsigned short* wbuf /* [2][PSZ] */,
    int tid, int l16, int half, int mt, int nq)
{
  v8f acc[8] = {};
  const int nKP = ktot / KP;
  issue_copy(gWblk, wbuf, PSZ, tid);
  wait_async0();
  __syncthreads();
  for (int kp = 0; kp < nKP; ++kp) {
    const unsigned short* cur = wbuf + (kp & 1) * PSZ;
    if (kp + 1 < nKP)   // overlap: DMA next panel while doing WMMAs on this one
      issue_copy(gWblk + (size_t)(kp + 1) * PSZ, wbuf + ((kp + 1) & 1) * PSZ, PSZ, tid);
    v16bf a = load_afrag(actIn + (mt * 16 + l16) * inStride + kp * KP, half);
#pragma unroll
    for (int t = 0; t < 8; ++t) {
      int ncol = nq * 128 + t * 16 + l16;
      acc[t] = wmma_bf16(a, load_bfrag(cur + ncol * KP, half), acc[t]);
    }
    wait_async0();        // own chunks of panel kp+1 arrived
    __syncthreads();      // publishes kp+1 to all; retires all readers of cur
  }
#pragma unroll
  for (int t = 0; t < 8; ++t) {
    int ncol = nq * 128 + t * 16 + l16;
    float bv = gB[ncol];
#pragma unroll
    for (int r = 0; r < 8; ++r) {
      float v = acc[t][r] + bv;
      if (elu_act) v = elu(v);
      actOut[(mt * 16 + half * 8 + r) * HDIM + ncol] = f2bf(v);
    }
  }
}

// ---------------- fallback trunk layer: sync fp32 loads + in-LDS conversion ----------------
__device__ __forceinline__ void dense_layer_sync(
    const unsigned short* actIn, int inStride, int ktot,
    const float* __restrict__ gW, const float* __restrict__ gB, bool elu_act,
    unsigned short* actOut, unsigned short* wts,
    int tid, int l16, int half, int mt, int nq)
{
  v8f acc[8] = {};
  const int nKP = ktot / KP;
  for (int kp = 0; kp < nKP; ++kp) {
    __syncthreads();
    for (int i = tid; i < KP * HDIM; i += NTHREADS) {
      int k = i >> 8;
      int n = i & (HDIM - 1);
      wts[n * KP + k] = f2bf(gW[(size_t)(kp * KP + k) * HDIM + n]);
    }
    if (kp + 1 < nKP)
      __builtin_prefetch(gW + (size_t)(kp + 1) * KP * HDIM + tid * 16, 0, 1);
    __syncthreads();
    v16bf a = load_afrag(actIn + (mt * 16 + l16) * inStride + kp * KP, half);
#pragma unroll
    for (int t = 0; t < 8; ++t) {
      int ncol = nq * 128 + t * 16 + l16;
      acc[t] = wmma_bf16(a, load_bfrag(wts + ncol * KP, half), acc[t]);
    }
  }
  __syncthreads();
#pragma unroll
  for (int t = 0; t < 8; ++t) {
    int ncol = nq * 128 + t * 16 + l16;
    float bv = gB[ncol];
#pragma unroll
    for (int r = 0; r < 8; ++r) {
      float v = acc[t][r] + bv;
      if (elu_act) v = elu(v);
      actOut[(mt * 16 + half * 8 + r) * HDIM + ncol] = f2bf(v);
    }
  }
}

// ---------------- fused ensemble-MLP sampler ----------------
template<bool PRE>
__global__ void __launch_bounds__(NTHREADS, 1)
bnn_ensemble_mlp(const float* __restrict__ x,
                 const float* __restrict__ W0, const float* __restrict__ b0,
                 const float* __restrict__ W1, const float* __restrict__ b1,
                 const float* __restrict__ W2, const float* __restrict__ b2,
                 const float* __restrict__ Wm, const float* __restrict__ bm,
                 const float* __restrict__ Ws, const float* __restrict__ bs,
                 const float* __restrict__ eps, float* __restrict__ out,
                 const unsigned short* __restrict__ cW0, const unsigned short* __restrict__ cW1,
                 const unsigned short* __restrict__ cW2, const unsigned short* __restrict__ cWm,
                 const unsigned short* __restrict__ cWs)
{
  __shared__ __attribute__((aligned(16))) unsigned short actA[MT * HDIM];  // 32 KB
  __shared__ __attribute__((aligned(16))) unsigned short actB[MT * HDIM];  // 32 KB
  __shared__ __attribute__((aligned(16))) unsigned short wbuf[2 * PSZ];    // 32 KB: panels / staging

  const int tid  = threadIdx.x;
  const int lane = tid & 31;
  const int wid  = tid >> 5;
  const int half = lane >> 4;
  const int l16  = lane & 15;
  const int mt   = wid >> 1;    // 0..3 : wave's 16-row M tile
  const int nq   = wid & 1;     // 0..1 : wave's 128-col N half

  const int w  = blockIdx.x & (WENS - 1);
  const int n0 = (blockIdx.x >> 8) * MT;

  // ---- async load of the X tile (64x32 fp32, 8 KB) into LDS staging (aliases wbuf) ----
  float* stage = (float*)wbuf;
  {
    const float* bx = x + ((size_t)n0 * WENS + (size_t)w) * XDIM;  // uniform base (SGPR)
#pragma unroll
    for (int it = 0; it < 2; ++it) {
      int c = it * NTHREADS + tid;        // 512 x 16B chunks
      int row = c >> 3;
      int c8  = c & 7;
      unsigned goff = (unsigned)(row * (WENS * XDIM * 4) + c8 * 16);
      unsigned loff = lds_off(stage) + (unsigned)((row * XDIM + c8 * 4) * 4);
      async_b128(loff, goff, bx);
    }
    wait_async0();
  }
  __syncthreads();
  for (int i = tid; i < MT * XDIM; i += NTHREADS) actA[i] = f2bf(stage[i]);
  __syncthreads();   // staging (wbuf) free for panel reuse

  // ---- fused trunk: h0, h1, h2 stay in LDS ----
  if (PRE) {
    dense_layer_pre(actA, XDIM, XDIM, cW0 + (size_t)w * 1 * PSZ,
                    b0 + (size_t)w * HDIM, true, actB, wbuf, tid, l16, half, mt, nq);
    dense_layer_pre(actB, HDIM, HDIM, cW1 + (size_t)w * 8 * PSZ,
                    b1 + (size_t)w * HDIM, true, actA, wbuf, tid, l16, half, mt, nq);
    dense_layer_pre(actA, HDIM, HDIM, cW2 + (size_t)w * 8 * PSZ,
                    b2 + (size_t)w * HDIM, true, actB, wbuf, tid, l16, half, mt, nq);
  } else {
    dense_layer_sync(actA, XDIM, XDIM, W0 + (size_t)w * XDIM * HDIM,
                     b0 + (size_t)w * HDIM, true, actB, wbuf, tid, l16, half, mt, nq);
    dense_layer_sync(actB, HDIM, HDIM, W1 + (size_t)w * HDIM * HDIM,
                     b1 + (size_t)w * HDIM, true, actA, wbuf, tid, l16, half, mt, nq);
    dense_layer_sync(actA, HDIM, HDIM, W2 + (size_t)w * HDIM * HDIM,
                     b2 + (size_t)w * HDIM, true, actB, wbuf, tid, l16, half, mt, nq);
  }

  // ---- heads: all 8 waves active (4 M-tiles x 2 N-tiles), both heads share A fragments ----
  const int hm = wid >> 1;   // 0..3
  const int hn = wid & 1;    // 0..1
  v8f am = {}, asd = {};

  if (PRE) {
    const unsigned short* gm = cWm + (size_t)w * 8 * HPSZ;
    const unsigned short* gs = cWs + (size_t)w * 8 * HPSZ;
    issue_copy(gm, wbuf, HPSZ, tid);
    issue_copy(gs, wbuf + HPSZ, HPSZ, tid);
    wait_async0();
    __syncthreads();
    for (int kp = 0; kp < HDIM / KP; ++kp) {
      const unsigned short* cur = wbuf + (kp & 1) * PSZ;
      if (kp + 1 < HDIM / KP) {
        unsigned short* nxt = wbuf + ((kp + 1) & 1) * PSZ;
        issue_copy(gm + (size_t)(kp + 1) * HPSZ, nxt, HPSZ, tid);
        issue_copy(gs + (size_t)(kp + 1) * HPSZ, nxt + HPSZ, HPSZ, tid);
      }
      v16bf a = load_afrag(actB + (hm * 16 + l16) * HDIM + kp * KP, half);
      am  = wmma_bf16(a, load_bfrag(cur + (hn * 16 + l16) * KP, half), am);
      asd = wmma_bf16(a, load_bfrag(cur + HPSZ + (hn * 16 + l16) * KP, half), asd);
      wait_async0();
      __syncthreads();
    }
  } else {
    const float* gWm = Wm + (size_t)w * HDIM * YDIM;
    const float* gWs = Ws + (size_t)w * HDIM * YDIM;
    unsigned short* wmT = wbuf + PSZ;
    unsigned short* wsT = wbuf + PSZ + HPSZ;
    for (int kp = 0; kp < HDIM / KP; ++kp) {
      __syncthreads();
      for (int i = tid; i < KP * YDIM; i += NTHREADS) {
        int k = i >> 5;
        int n = i & (YDIM - 1);
        wmT[n * KP + k] = f2bf(gWm[(size_t)(kp * KP + k) * YDIM + n]);
        wsT[n * KP + k] = f2bf(gWs[(size_t)(kp * KP + k) * YDIM + n]);
      }
      __syncthreads();
      v16bf a = load_afrag(actB + (hm * 16 + l16) * HDIM + kp * KP, half);
      am  = wmma_bf16(a, load_bfrag(wmT + (hn * 16 + l16) * KP, half), am);
      asd = wmma_bf16(a, load_bfrag(wsT + (hn * 16 + l16) * KP, half), asd);
    }
  }

  // ---- sampling epilogue: out = mean + sqrt(exp(s/20 + 1)) * eps ----
  {
    int ncol = hn * 16 + l16;
    float bmv = bm[(size_t)w * YDIM + ncol];
    float bsv = bs[(size_t)w * YDIM + ncol];
#pragma unroll
    for (int r = 0; r < 8; ++r) {
      int row = hm * 16 + half * 8 + r;
      size_t idx = (((size_t)(n0 + row)) * WENS + (size_t)w) * YDIM + ncol;
      float mean = am[r] + bmv;
      float ls   = (asd[r] + bsv) * 0.05f + 1.0f;        // log_std
      out[idx] = mean + __expf(0.5f * ls) * eps[idx];    // sqrt(exp(ls)) = exp(ls/2)
    }
  }
}

extern "C" void kernel_launch(void* const* d_in, const int* in_sizes, int n_in,
                              void* d_out, int out_size, void* d_ws, size_t ws_size,
                              hipStream_t stream) {
  const float* x   = (const float*)d_in[0];
  const float* W0  = (const float*)d_in[1];
  const float* b0  = (const float*)d_in[2];
  const float* W1  = (const float*)d_in[3];
  const float* b1  = (const float*)d_in[4];
  const float* W2  = (const float*)d_in[5];
  const float* b2  = (const float*)d_in[6];
  const float* Wm  = (const float*)d_in[7];
  const float* bm  = (const float*)d_in[8];
  const float* Ws  = (const float*)d_in[9];
  const float* bs  = (const float*)d_in[10];
  const float* eps = (const float*)d_in[11];
  float* out = (float*)d_out;

  const size_t nW0 = (size_t)WENS * XDIM * HDIM;   // elements
  const size_t nW1 = (size_t)WENS * HDIM * HDIM;
  const size_t nWm = (size_t)WENS * HDIM * YDIM;
  const size_t needBytes = (nW0 + 2 * nW1 + 2 * nWm) * sizeof(unsigned short);

  dim3 grid(WENS * (NPTS / MT));   // 256 members x 16 row tiles
  if (ws_size >= needBytes) {
    unsigned short* cW0 = (unsigned short*)d_ws;
    unsigned short* cW1 = cW0 + nW0;
    unsigned short* cW2 = cW1 + nW1;
    unsigned short* cWm = cW2 + nW1;
    unsigned short* cWs = cWm + nWm;
    hipLaunchKernelGGL((convert_weights<8>), dim3(WENS * (XDIM / KP)), dim3(NTHREADS), 0, stream, W0, cW0, XDIM / KP);
    hipLaunchKernelGGL((convert_weights<8>), dim3(WENS * (HDIM / KP)), dim3(NTHREADS), 0, stream, W1, cW1, HDIM / KP);
    hipLaunchKernelGGL((convert_weights<8>), dim3(WENS * (HDIM / KP)), dim3(NTHREADS), 0, stream, W2, cW2, HDIM / KP);
    hipLaunchKernelGGL((convert_weights<5>), dim3(WENS * (HDIM / KP)), dim3(NTHREADS), 0, stream, Wm, cWm, HDIM / KP);
    hipLaunchKernelGGL((convert_weights<5>), dim3(WENS * (HDIM / KP)), dim3(NTHREADS), 0, stream, Ws, cWs, HDIM / KP);
    hipLaunchKernelGGL((bnn_ensemble_mlp<true>), grid, dim3(NTHREADS), 0, stream,
                       x, W0, b0, W1, b1, W2, b2, Wm, bm, Ws, bs, eps, out,
                       cW0, cW1, cW2, cWm, cWs);
  } else {
    hipLaunchKernelGGL((bnn_ensemble_mlp<false>), grid, dim3(NTHREADS), 0, stream,
                       x, W0, b0, W1, b1, W2, b2, Wm, bm, Ws, bs, eps, out,
                       (const unsigned short*)nullptr, (const unsigned short*)nullptr,
                       (const unsigned short*)nullptr, (const unsigned short*)nullptr,
                       (const unsigned short*)nullptr);
  }
  (void)in_sizes; (void)n_in; (void)out_size;
}
